// PortHamiltonianDynamics_8727373545727
// MI455X (gfx1250) — matrix operand order repaired and checked by reading, not compiled, not run on hardware
//
#include <hip/hip_runtime.h>

#define DIM   64
#define HID   128
#define BATCH 16384
#define WAVES 4
#define LDZ   72     // z tile row stride (halves), 144B rows, 16B aligned
#define LDB   136    // activation/stage row stride (halves), 272B rows, 16B aligned, 68-bank rotation

#define STAGE_ROWS 64
#define STAGE_USH  (STAGE_ROWS * LDB)          // one stage buffer (64 rows x 136 halves)
// per-wave LDS (ushort units): Z(16*72) + P1/P2/X/Db(4*16*136) + 48 floats
#define WAVE_USH (16*LDZ + 4*16*LDB + 96)

#if __has_builtin(__builtin_amdgcn_tensor_load_to_lds)
#define HAVE_TDM 1
#else
#define HAVE_TDM 0
#endif

typedef __bf16 v16bf __attribute__((ext_vector_type(16)));
typedef float  v8f   __attribute__((ext_vector_type(8)));
typedef unsigned int u32x4 __attribute__((ext_vector_type(4)));
typedef int          i32x8 __attribute__((ext_vector_type(8)));
typedef int          i32x4 __attribute__((ext_vector_type(4)));

// ---------------- bf16 helpers (raw ushort storage) ----------------
static __device__ __forceinline__ unsigned short f32_to_bf16_raw(float f){
  union { float f; unsigned u; } c; c.f = f;
  unsigned u = c.u;
  unsigned r = (u + 0x7FFFu + ((u >> 16) & 1u)) >> 16;   // RNE
  return (unsigned short)r;
}
static __device__ __forceinline__ float bf16_raw_to_f32(unsigned short s){
  union { unsigned u; float f; } c; c.u = ((unsigned)s) << 16;
  return c.f;
}

static __device__ __forceinline__ float sigmoidf_(float x){ return 1.0f / (1.0f + __expf(-x)); }
static __device__ __forceinline__ float siluf_(float x){ return x * sigmoidf_(x); }
static __device__ __forceinline__ float dsiluf_(float x){
  float s = sigmoidf_(x); return s * (1.0f + x * (1.0f - s));
}
static __device__ __forceinline__ float softplusf_(float x){
  return (x > 20.0f) ? x : log1pf(__expf(x));
}

static __device__ __forceinline__ int lane_id(){ return (int)(threadIdx.x & 31u); }

// ---------------- WMMA operand loaders ----------------
// A operand 16x32 bf16 from LDS row-major [16][lda]: lanes 0-15 row=lane hold
// K=kOff+{0..15}; lanes 16-31 row=lane-16 hold K=kOff+{16..31}.
static __device__ __forceinline__ v16bf load_a_lds(const unsigned short* base, int lda, int kOff){
  const int l = lane_id();
  const int row = l & 15;
  const int kb = kOff + ((l >> 4) << 4);
  const uint4* q = (const uint4*)(base + row * lda + kb);
  union { v16bf v; uint4 q[2]; } r;
  r.q[0] = q[0];
  r.q[1] = q[1];
  return r.v;
}
// B operand 32x16 bf16 from global weights stored transposed [N][K] row-major:
// lane holds one contiguous 32B chunk of its column's K-run.
static __device__ __forceinline__ v16bf load_b_glb(const unsigned short* w, int ldk, int n0, int kOff){
  const int l = lane_id();
  const uint4* q = (const uint4*)(w + (unsigned)(n0 + (l & 15)) * (unsigned)ldk
                                    + kOff + ((l >> 4) << 4));
  union { v16bf v; uint4 q[2]; } r;
  r.q[0] = q[0];
  r.q[1] = q[1];
  return r.v;
}
// B operand from LDS weight stage (row stride LDB halves -> conflict-free banks)
static __device__ __forceinline__ v16bf load_b_lds(const unsigned short* wst, int n0, int kOff){
  const int l = lane_id();
  const uint4* q = (const uint4*)(wst + (n0 + (l & 15)) * LDB + kOff + ((l >> 4) << 4));
  union { v16bf v; uint4 q[2]; } r;
  r.q[0] = q[0];
  r.q[1] = q[1];
  return r.v;
}

static __device__ __forceinline__ v8f wmma_bf16(v16bf a, v16bf b, v8f c){
  return __builtin_amdgcn_wmma_f32_16x16x32_bf16(false, a, false, b, (short)0, c, false, false);
}

template<int NK>
static __device__ __forceinline__ v8f tile_mm_glb(const v16bf* a, const unsigned short* W, int n0){
  v8f c = {0.f,0.f,0.f,0.f,0.f,0.f,0.f,0.f};
#pragma unroll
  for (int k = 0; k < NK; ++k)
    c = wmma_bf16(a[k], load_b_glb(W, NK*32, n0, k*32), c);
  return c;
}
static __device__ __forceinline__ v8f tile_mm_lds(const v16bf* a, const unsigned short* Wst, int n0){
  v8f c = {0.f,0.f,0.f,0.f,0.f,0.f,0.f,0.f};
#pragma unroll
  for (int k = 0; k < 4; ++k)
    c = wmma_bf16(a[k], load_b_lds(Wst, n0, k*32), c);
  return c;
}

// ---------------- TDM weight staging: 64 rows x 128 halves -> LDS ----------------
static __device__ __forceinline__ void stage_issue(const unsigned short* gsrc,
                                                   unsigned ldsByteOff,
                                                   unsigned short* ldst,
                                                   int wave, int tid)
{
#if HAVE_TDM
  (void)ldst; (void)tid;
  if (wave == 0){
    const unsigned long long ga = (unsigned long long)(uintptr_t)gsrc;
    u32x4 g0;
    g0[0] = 1u;                                          // count=1, user descriptor
    g0[1] = ldsByteOff;                                  // lds_addr (bytes)
    g0[2] = (unsigned)(ga & 0xFFFFFFFFu);                // global_addr[31:0]
    g0[3] = (unsigned)((ga >> 32) & 0x1FFFFFFu)          // global_addr[56:32]
          | (2u << 30);                                  // type = 2 ("image")
    i32x8 g1;
    g1[0] = (1 << 16)                                    // data_size = 2 bytes
          | (1 << 20)                                    // pad_enable
          | (5 << 22)                                    // pad_interval: every 64 DWORDs
          | (3 << 25);                                   // pad_amount: 4 DWORDs (16B)
    g1[1] = (HID << 16);                                 // tensor_dim0 = 128 (lo16)
    g1[2] = (int)(4096u << 16);                          // tensor_dim1 = 4096 (lo16)
    g1[3] = (HID << 16);                                 // tile_dim0 = 128
    g1[4] = STAGE_ROWS;                                  // tile_dim1 = 64, tile_dim2 = 0
    g1[5] = HID;                                         // tensor_dim0_stride = 128
    g1[6] = 0;
    g1[7] = 0;
    const i32x4 gz = {0,0,0,0};
#if defined(__clang_major__) && (__clang_major__ >= 23)
    const i32x8 gz8 = {0,0,0,0,0,0,0,0};
    __builtin_amdgcn_tensor_load_to_lds(g0, g1, gz, gz, gz8, 0);
#else
    __builtin_amdgcn_tensor_load_to_lds(g0, g1, gz, gz, 0);
#endif
  }
#else
  (void)ldsByteOff; (void)wave;
  for (int i = tid; i < STAGE_ROWS * 16; i += WAVES*32){   // 16 uint4 per row
    const int row = i >> 4, c8 = (i & 15) * 8;
    *(uint4*)(ldst + row*LDB + c8) = *(const uint4*)(gsrc + row*HID + c8);
  }
#endif
}
static __device__ __forceinline__ void stage_wait(int wave, bool more_in_flight)
{
#if HAVE_TDM
  if (wave == 0){
    if (more_in_flight) __builtin_amdgcn_s_wait_tensorcnt(1);
    else                __builtin_amdgcn_s_wait_tensorcnt(0);
  }
#else
  (void)wave; (void)more_in_flight;
#endif
}
static __device__ __forceinline__ unsigned lds_byte_off_of_dyn(unsigned dynOff){
  return __builtin_amdgcn_groupstaticsize() + dynOff;     // dynamic LDS starts after static
}

// ---------------- GEMM layer: out[16,N] = act(A[16,K] @ Wt + bias) ----------------
// act: 0 = none, 1 = silu, 2 = tanh. Optionally stores pre-activation (bf16).
template<int NK>
static __device__ __forceinline__ void gemm_layer(
    const unsigned short* A, int lda,
    const unsigned short* W, const float* bias, int N,
    unsigned short* out, unsigned short* pre, int act)
{
  const int l   = lane_id();
  const int col = l & 15;
  const int hl  = l >> 4;
  v16bf a[NK];
#pragma unroll
  for (int k = 0; k < NK; ++k) a[k] = load_a_lds(A, lda, k*32);
  for (int n0 = 0; n0 < N; n0 += 16){
    v8f c = tile_mm_glb<NK>(a, W, n0);
    const float bv = bias[n0 + col];
#pragma unroll
    for (int v = 0; v < 8; ++v){
      const int b = v + (hl << 3);
      const float x = c[v] + bv;
      if (pre) pre[b*LDB + n0 + col] = f32_to_bf16_raw(x);
      float y = x;
      if (act == 1) y = siluf_(x);
      else if (act == 2) y = tanhf(x);
      out[b*LDB + n0 + col] = f32_to_bf16_raw(y);
    }
  }
}

// ---------------- scalar net fwd+bwd: H, gs, and gz in C-layout regs ----------------
static __device__ __forceinline__ void scalar_net(
    const unsigned short* Z,
    const unsigned short* w1t, const float* b1,
    const unsigned short* w2t, const float* b2,
    const unsigned short* w2p, const unsigned short* w1p,
    const float* w3, const float* b3p,
    unsigned short* P1, unsigned short* P2, unsigned short* X, unsigned short* Db,
    float* Hout, float* gsv, v8f* g)
{
  const int l   = lane_id();
  const int col = l & 15;
  const int hl  = l >> 4;

  gemm_layer<2>(Z, LDZ, w1t, b1, HID, X, P1, 1);    // pre1 -> P1, h1 -> X
  gemm_layer<4>(X, LDB, w2t, b2, HID, Db, P2, 1);   // pre2 -> P2, h2 -> Db

  // s = h2 @ w3 + b3; H = softplus(s); gs = sigmoid(s)
  {
    const int row = col;
    const int k0  = hl << 6;
    float sacc = 0.f;
    for (int k = 0; k < 64; ++k)
      sacc += bf16_raw_to_f32(Db[row*LDB + k0 + k]) * w3[k0 + k];
    sacc += __shfl_xor(sacc, 16);
    if (l < 16){
      const float sv = sacc + b3p[0];
      Hout[row] = softplusf_(sv);
      gsv[row]  = sigmoidf_(sv);
    }
  }

  // d_pre2 = gs * w3 * silu'(pre2)  -> Db
  for (int i = l; i < 16*HID; i += 32){
    const int b = i >> 7, k = i & 127;
    const float pre2 = bf16_raw_to_f32(P2[b*LDB + k]);
    Db[b*LDB + k] = f32_to_bf16_raw(gsv[b] * w3[k] * dsiluf_(pre2));
  }

  // d_pre1 = (d_pre2 @ W2^T) * silu'(pre1) -> X  (B storage = plain W2 [128][128])
  {
    v16bf a[4];
#pragma unroll
    for (int k = 0; k < 4; ++k) a[k] = load_a_lds(Db, LDB, k*32);
    for (int n0 = 0; n0 < HID; n0 += 16){
      v8f c = tile_mm_glb<4>(a, w2p, n0);
#pragma unroll
      for (int v = 0; v < 8; ++v){
        const int b = v + (hl << 3);
        const float pre1 = bf16_raw_to_f32(P1[b*LDB + n0 + col]);
        X[b*LDB + n0 + col] = f32_to_bf16_raw(c[v] * dsiluf_(pre1));
      }
    }
  }

  // gz = d_pre1 @ W1^T  (B storage = plain W1 [64][128]) -> C-layout regs
  {
    v16bf a[4];
#pragma unroll
    for (int k = 0; k < 4; ++k) a[k] = load_a_lds(X, LDB, k*32);
#pragma unroll
    for (int s = 0; s < 4; ++s)
      g[s] = tile_mm_glb<4>(a, w1p, s*16);
  }
}

// ---------------- streaming pass with TDM-staged weights ----------------
// Tile (r, c0): T[b, c] = (h @ W2t)[b, r*64 + c0*16 + c] + bias  (Mat[b][r][c])
// T2: t2acc[c0][v] += t2s * T * src[b][r]   (column accumulation, C-layout regs)
// T1: t1acc[r/16][v] += t1s * sum_c T*src[b][c0*16+c]  (row accumulation, lane-reduced)
template<bool T2, bool T1>
static __device__ __forceinline__ void stream_pass(
    const v16bf a[4],
    const unsigned short* Wglb, unsigned short* Wstage, const float* bias,
    const v8f* src, v8f* t2acc, float t2s, v8f* t1acc, float t1s,
    int wave, int tid)
{
  const int l   = lane_id();
  const int col = l & 15;
  stage_issue(Wglb, lds_byte_off_of_dyn(0), Wstage, wave, tid);
#pragma unroll
  for (int s = 0; s < 4; ++s){
    for (int rr = 0; rr < 16; ++rr){
      const int r = s*16 + rr;
      const unsigned short* cur = Wstage + (r & 1) * STAGE_USH;
      if (r < 63){
        const int nb = (r + 1) & 1;
        stage_issue(Wglb + (size_t)(r + 1) * STAGE_ROWS * HID,
                    lds_byte_off_of_dyn((unsigned)nb * STAGE_USH * 2u),
                    Wstage + nb * STAGE_USH, wave, tid);
      }
      stage_wait(wave, r < 63);
      __syncthreads();                 // stage buffer r ready for all waves
      float br[8];
      if (T2){
#pragma unroll
        for (int v = 0; v < 8; ++v) br[v] = __shfl(src[s][v], rr, 16);
      }
#pragma unroll
      for (int c0 = 0; c0 < 4; ++c0){
        v8f c = tile_mm_lds(a, cur, c0*16);
        const float bv = bias[r*64 + c0*16 + col];
#pragma unroll
        for (int v = 0; v < 8; ++v){
          const float T = c[v] + bv;
          if (T2) t2acc[c0][v] += t2s * (T * br[v]);
          if (T1){
            float p2 = T * src[c0][v];
            p2 += __shfl_xor(p2, 1);
            p2 += __shfl_xor(p2, 2);
            p2 += __shfl_xor(p2, 4);
            p2 += __shfl_xor(p2, 8);
            if (col == rr) t1acc[s][v] += t1s * p2;
          }
        }
      }
      __syncthreads();                 // all waves done before buffer reuse
    }
  }
}

// ---------------- params ----------------
struct Params {
  const float *z;
  const float *Hb1, *Hb2, *Hw3, *Hb3;
  const float *Db1, *Db2, *Dw3, *Db3;
  const float *Jb1, *Jb2, *Bb1, *Bb2, *ub1, *ub2;
  const float *dt_logit, *energy;
  const unsigned short *Hw1t, *Hw1p, *Hw2t, *Hw2p;
  const unsigned short *Dw1t, *Dw1p, *Dw2t, *Dw2p;
  const unsigned short *Jw1t, *Jw2t, *Bw1t, *Bw2t, *uw1t, *uw2t;
  float *out;
};

// ---------------- main kernel: one wave = 16 batch rows ----------------
__global__ __launch_bounds__(WAVES*32) void ph_kernel(Params p)
{
  extern __shared__ __align__(16) unsigned short smem[];
  const int tid  = (int)threadIdx.x;
  const int wave = tid >> 5;
  const int l    = lane_id();
  const int col  = l & 15;
  const int hl   = l >> 4;

  unsigned short* Wstage = smem;                       // 2 x [64][LDB] staged weights (shared)
  unsigned short* base = smem + 2*STAGE_USH + (size_t)wave * WAVE_USH;
  unsigned short* Z  = base;                 // [16][LDZ] bf16 z tile
  unsigned short* P1 = Z  + 16*LDZ;
  unsigned short* P2 = P1 + 16*LDB;
  unsigned short* X  = P2 + 16*LDB;
  unsigned short* Db = X  + 16*LDB;
  float* fH   = (float*)(Db + 16*LDB);
  float* fG   = fH + 16;
  float* fDum = fG + 16;

  const int tileB = ((int)blockIdx.x * WAVES + wave) * 16;

  for (int i = l; i < 16*DIM; i += 32){
    const int b = i >> 6, c = i & 63;
    Z[b*LDZ + c] = f32_to_bf16_raw(p.z[(size_t)(tileB + b)*DIM + c]);
  }

  v8f ghc[4], gdc[4], acc[4], tC[4];
  {
    const v8f zed = {0.f,0.f,0.f,0.f,0.f,0.f,0.f,0.f};
#pragma unroll
    for (int s = 0; s < 4; ++s){ acc[s] = zed; tC[s] = zed; ghc[s] = zed; gdc[s] = zed; }
  }

  scalar_net(Z, p.Hw1t, p.Hb1, p.Hw2t, p.Hb2, p.Hw2p, p.Hw1p, p.Hw3, p.Hb3,
             P1, P2, X, Db, fH, fG, ghc);
  scalar_net(Z, p.Dw1t, p.Db1, p.Dw2t, p.Db2, p.Dw2p, p.Dw1p, p.Dw3, p.Db3,
             P1, P2, X, Db, fDum, fG, gdc);

  // ---- J path: cons = A gH - A^T gH, one streamed pass over A tiles ----
  gemm_layer<2>(Z, LDZ, p.Jw1t, p.Jb1, HID, X, nullptr, 1);
  {
    v16bf a[4];
#pragma unroll
    for (int k = 0; k < 4; ++k) a[k] = load_a_lds(X, LDB, k*32);
    stream_pass<true, true>(a, p.Jw2t, Wstage, p.Jb2, ghc, acc, -1.f, acc, +1.f, wave, tid);
  }

  // ---- B path: diss = -Bm (Bm^T gD), two streamed passes (R never materialized) ----
  gemm_layer<2>(Z, LDZ, p.Bw1t, p.Bb1, HID, X, nullptr, 1);
  {
    v16bf a[4];
#pragma unroll
    for (int k = 0; k < 4; ++k) a[k] = load_a_lds(X, LDB, k*32);
    stream_pass<true, false>(a, p.Bw2t, Wstage, p.Bb2, gdc, tC, +1.f, nullptr, 0.f, wave, tid); // t = Bm^T gD
    stream_pass<false, true>(a, p.Bw2t, Wstage, p.Bb2, tC, nullptr, 0.f, acc, -1.f, wave, tid); // acc -= Bm t
  }

  // ---- u path + pump + integrate ----
  gemm_layer<2>(Z, LDZ, p.uw1t, p.ub1, HID, X, nullptr, 2);
  const float dt = sigmoidf_(p.dt_logit[0]) * 0.1f;   // sigmoid(dt_logit) * DT_MAX
  const float E  = p.energy[0];
  {
    v16bf a[4];
#pragma unroll
    for (int k = 0; k < 4; ++k) a[k] = load_a_lds(X, LDB, k*32);
#pragma unroll
    for (int s = 0; s < 4; ++s){
      v8f c = tile_mm_glb<4>(a, p.uw2t, s*16);
      const float bv = p.ub2[s*16 + col];
#pragma unroll
      for (int v = 0; v < 8; ++v){
        const int b = v + (hl << 3);
        const float uval = c[v] + bv;
        const float dz = acc[s][v] + 0.1f * (E - fH[b]) * uval;
        const size_t gi = (size_t)(tileB + b)*DIM + s*16 + col;
        p.out[gi] = p.z[gi] + dt * dz;
      }
    }
  }
}

// ---------------- weight conversion: f32 -> bf16, optional transpose ----------------
__global__ void conv_w(const float* __restrict__ src, unsigned short* __restrict__ dst,
                       int fi, int fo, int transpose)
{
  const int i = (int)(blockIdx.x * blockDim.x + threadIdx.x);
  const int n = fi * fo;
  if (i >= n) return;
  const int r = i / fo, c = i - r * fo;
  const unsigned short v = f32_to_bf16_raw(src[i]);
  if (transpose) dst[(size_t)c * fi + r] = v;
  else           dst[i] = v;
}

// ---------------- workspace layout (ushort elements) ----------------
enum : size_t {
  O_Hw1t = 0,
  O_Hw1p = O_Hw1t + 128*64,
  O_Hw2t = O_Hw1p + 64*128,
  O_Hw2p = O_Hw2t + 128*128,
  O_Dw1t = O_Hw2p + 128*128,
  O_Dw1p = O_Dw1t + 128*64,
  O_Dw2t = O_Dw1p + 64*128,
  O_Dw2p = O_Dw2t + 128*128,
  O_Jw1t = O_Dw2p + 128*128,
  O_Jw2t = O_Jw1t + 128*64,
  O_Bw1t = O_Jw2t + (size_t)4096*128,
  O_Bw2t = O_Bw1t + 128*64,
  O_uw1t = O_Bw2t + (size_t)4096*128,
  O_uw2t = O_uw1t + 128*64,
  O_END  = O_uw2t + 64*128
};

extern "C" void kernel_launch(void* const* d_in, const int* in_sizes, int n_in,
                              void* d_out, int out_size, void* d_ws, size_t ws_size,
                              hipStream_t stream)
{
  (void)in_sizes; (void)n_in; (void)out_size; (void)ws_size;
  unsigned short* ws = (unsigned short*)d_ws;

  auto conv = [&](int idx, size_t off, int fi, int fo, int tr){
    const int n = fi * fo, thr = 256, blk = (n + thr - 1) / thr;
    conv_w<<<dim3(blk), dim3(thr), 0, stream>>>((const float*)d_in[idx], ws + off, fi, fo, tr);
  };
  conv( 1, O_Hw1t,  64,  128, 1);  conv( 1, O_Hw1p,  64,  128, 0);
  conv( 3, O_Hw2t, 128,  128, 1);  conv( 3, O_Hw2p, 128,  128, 0);
  conv( 7, O_Dw1t,  64,  128, 1);  conv( 7, O_Dw1p,  64,  128, 0);
  conv( 9, O_Dw2t, 128,  128, 1);  conv( 9, O_Dw2p, 128,  128, 0);
  conv(13, O_Jw1t,  64,  128, 1);
  conv(15, O_Jw2t, 128, 4096, 1);
  conv(17, O_Bw1t,  64,  128, 1);
  conv(19, O_Bw2t, 128, 4096, 1);
  conv(21, O_uw1t,  64,  128, 1);
  conv(23, O_uw2t, 128,   64, 1);

  Params p;
  p.z        = (const float*)d_in[0];
  p.Hb1      = (const float*)d_in[2];
  p.Hb2      = (const float*)d_in[4];
  p.Hw3      = (const float*)d_in[5];
  p.Hb3      = (const float*)d_in[6];
  p.Db1      = (const float*)d_in[8];
  p.Db2      = (const float*)d_in[10];
  p.Dw3      = (const float*)d_in[11];
  p.Db3      = (const float*)d_in[12];
  p.Jb1      = (const float*)d_in[14];
  p.Jb2      = (const float*)d_in[16];
  p.Bb1      = (const float*)d_in[18];
  p.Bb2      = (const float*)d_in[20];
  p.ub1      = (const float*)d_in[22];
  p.ub2      = (const float*)d_in[24];
  p.dt_logit = (const float*)d_in[25];
  p.energy   = (const float*)d_in[26];
  p.Hw1t = ws + O_Hw1t;  p.Hw1p = ws + O_Hw1p;
  p.Hw2t = ws + O_Hw2t;  p.Hw2p = ws + O_Hw2p;
  p.Dw1t = ws + O_Dw1t;  p.Dw1p = ws + O_Dw1p;
  p.Dw2t = ws + O_Dw2t;  p.Dw2p = ws + O_Dw2p;
  p.Jw1t = ws + O_Jw1t;  p.Jw2t = ws + O_Jw2t;
  p.Bw1t = ws + O_Bw1t;  p.Bw2t = ws + O_Bw2t;
  p.uw1t = ws + O_uw1t;  p.uw2t = ws + O_uw2t;
  p.out  = (float*)d_out;

  const size_t shmem = (size_t)(2*STAGE_USH + WAVES * WAVE_USH) * sizeof(unsigned short);
  ph_kernel<<<dim3(BATCH / (WAVES * 16)), dim3(WAVES * 32), shmem, stream>>>(p);
}
